// EdgeAwareMultiHeadAttention_68667937129203
// MI455X (gfx1250) — matrix operand order repaired and checked by reading, not compiled, not run on hardware
//
#include <hip/hip_runtime.h>
#include <hip/hip_bf16.h>
#include <math.h>

// Problem dimensions (fixed by the reference).
#define BGRAPH 256
#define NPG    128
#define NHEAD  8
#define DHEAD  32
#define HIDDIM 256
#define NEDGE  524288
#define NGTYPE 20
#define EFEAT  8
#define SCALE  0.17677669529663687f   // 1/sqrt(32)

typedef __attribute__((ext_vector_type(16))) _Float16 v16h;
typedef __attribute__((ext_vector_type(8)))  _Float16 v8h;
typedef __attribute__((ext_vector_type(4)))  _Float16 v4h;
typedef __attribute__((ext_vector_type(8)))  float    v8f;

// ---------------------------------------------------------------------------
// WMMA fragment helpers (wave32, V_WMMA_F32_16X16X32_F16), A/B use contiguous
// k = 16*(l/16)+s (same order in A and B, not the ISA table order).  All loads are lane-contiguous & 16B/32B
// aligned so they lower to ds_load_b128 / global_load_b128.
// ---------------------------------------------------------------------------

// A-matrix 16x32 f16 (row-major source, row stride ld halves):
// lane l -> M = l&15, K = [kb, kb+16), kb = (l<16)?0:16.
__device__ __forceinline__ v16h frag_a(const _Float16* a00, int ld, int lane) {
  const int m  = lane & 15;
  const int kb = (lane & 16) ? 16 : 0;
  const _Float16* r = a00 + m * ld + kb;
  v8h lo = *(const v8h*)(r);
  v8h hi = *(const v8h*)(r + 8);
  return __builtin_shufflevector(lo, hi, 0, 1, 2, 3, 4, 5, 6, 7,
                                 8, 9, 10, 11, 12, 13, 14, 15);
}

// B-matrix 32x16 f16 from a TRANSPOSED source t00 (t00[n*ld + k] == B[k][n]):
// lane l -> N = l&15, K = [kb, kb+16), kb = (l<16)?0:16.  One 32B load.
__device__ __forceinline__ v16h frag_b_t(const _Float16* t00, int ld, int lane) {
  const int n  = lane & 15;
  const int kb = (lane & 16) ? 16 : 0;
  return *(const v16h*)(t00 + n * ld + kb);
}

// C/D 16x16 f32: VGPR j, lane l -> row = j + (l>=16 ? 8 : 0), col = l&15.

// ---------------------------------------------------------------------------
// Kernel 0: transpose + convert a 256x256 f32 weight to f16 [out][in].
// ---------------------------------------------------------------------------
__global__ void transpose_w_kernel(const float* __restrict__ src,
                                   _Float16* __restrict__ dst) {
  const int idx = blockIdx.x * blockDim.x + threadIdx.x;   // 65536 threads
  const int k = idx >> 8;        // input row
  const int o = idx & 255;       // output column
  dst[o * HIDDIM + k] = (_Float16)src[k * HIDDIM + o];
}

// ---------------------------------------------------------------------------
// Kernel 1: zero the bias workspace (B*H*NPG*NPG f32 = 128 MiB, L2-resident)
// ---------------------------------------------------------------------------
__global__ void fill_zero_kernel(float4* p, long long n4) {
  long long i = (long long)blockIdx.x * blockDim.x + threadIdx.x;
  const long long stride = (long long)gridDim.x * blockDim.x;
  const float4 z = make_float4(0.f, 0.f, 0.f, 0.f);
  for (; i < n4; i += stride) p[i] = z;
}

// ---------------------------------------------------------------------------
// Kernel 2: per-edge bias = gate_table[type] + edge_attr @ We + be,
// scattered fwd + reverse (skip self-loop reverse) with f32 atomics.
// ---------------------------------------------------------------------------
__global__ void edge_bias_kernel(const int* __restrict__ ei,
                                 const float* __restrict__ eattr,
                                 const int* __restrict__ etype,
                                 const float* __restrict__ gate_table,
                                 const float* __restrict__ We,
                                 const float* __restrict__ be,
                                 float* __restrict__ bias) {
  const int e = blockIdx.x * blockDim.x + threadIdx.x;
  if (e >= NEDGE) return;
  const int src = ei[e];
  const int dst = ei[NEDGE + e];
  const int b   = src >> 7;          // NPG = 128
  const int sl  = src & (NPG - 1);
  const int dl  = dst & (NPG - 1);
  const int t   = etype[e];

  float tot[NHEAD];
#pragma unroll
  for (int h = 0; h < NHEAD; ++h) tot[h] = gate_table[t * NHEAD + h] + be[h];
#pragma unroll
  for (int f = 0; f < EFEAT; ++f) {
    const float a = eattr[e * EFEAT + f];
#pragma unroll
    for (int h = 0; h < NHEAD; ++h) tot[h] += a * We[f * NHEAD + h];
  }
  const long long base = (long long)b * NHEAD * NPG * NPG;
#pragma unroll
  for (int h = 0; h < NHEAD; ++h)
    atomicAdd(&bias[base + ((long long)h * NPG + sl) * NPG + dl], tot[h]);
  if (sl != dl) {
#pragma unroll
    for (int h = 0; h < NHEAD; ++h)
      atomicAdd(&bias[base + ((long long)h * NPG + dl) * NPG + sl], tot[h]);
  }
}

// ---------------------------------------------------------------------------
// Kernel 3: fused per-(graph, head) attention tile.
// 256 threads = 8 wave32; 184 KiB dynamic LDS (<= 320 KiB/WGP).
// ---------------------------------------------------------------------------

// 128x32 projection: sOut = sX @ WT_slice^T + bvec.  B fragments come straight
// from the pre-transposed f16 weight in global memory (L2-hot).
__device__ __forceinline__ void gemm_proj_128x32(const _Float16* sX,
                                                 const _Float16* __restrict__ WT,
                                                 const float* __restrict__ bvec,
                                                 int h, _Float16* sOut,
                                                 bool store_transposed,
                                                 int wave, int lane) {
#pragma unroll
  for (int i = 0; i < 2; ++i) {
    const int blk  = wave * 2 + i;       // 16 blocks: 8 row-tiles x 2 col-tiles
    const int row0 = (blk >> 1) * 16;
    const int col0 = (blk & 1) * 16;
    const _Float16* bbase = WT + (h * DHEAD + col0) * HIDDIM;
    v8f acc = {};
#pragma unroll
    for (int k0 = 0; k0 < HIDDIM; k0 += 32) {
      v16h a = frag_a(sX + row0 * HIDDIM + k0, HIDDIM, lane);
      v16h b = frag_b_t(bbase + k0, HIDDIM, lane);
      acc = __builtin_amdgcn_wmma_f32_16x16x32_f16(false, a, false, b,
                                                   (short)0, acc, false, false);
    }
    const int n  = lane & 15;
    const int rb = (lane & 16) ? 8 : 0;
    const float bb = bvec[h * DHEAD + col0 + n];
    if (!store_transposed) {
#pragma unroll
      for (int j = 0; j < 8; ++j)
        sOut[(row0 + rb + j) * DHEAD + col0 + n] = (_Float16)(acc[j] + bb);
    } else {
#pragma unroll
      for (int j = 0; j < 8; ++j)
        sOut[(col0 + n) * NPG + row0 + rb + j] = (_Float16)(acc[j] + bb);
    }
  }
}

__global__ __launch_bounds__(256)
void attention_tile_kernel(const float* __restrict__ x,
                           const _Float16* __restrict__ WqT,
                           const _Float16* __restrict__ WkT,
                           const _Float16* __restrict__ WvT,
                           const float* __restrict__ bq,
                           const float* __restrict__ bk,
                           const float* __restrict__ bv,
                           const float* __restrict__ bias,
                           float* __restrict__ attn_out) {
  extern __shared__ char smem[];
  _Float16* sX  = (_Float16*)(smem);             //  64 KiB [128][256] row-major
  _Float16* sQ  = (_Float16*)(smem + 65536);     //   8 KiB [128][32]  row-major
  _Float16* sK  = (_Float16*)(smem + 73728);     //   8 KiB [128][32]  row-major
  _Float16* sVt = (_Float16*)(smem + 81920);     //   8 KiB [32][128]  transposed
  float*    sS  = (float*)   (smem + 90112);     //  64 KiB [128][128] f32
  _Float16* sP  = (_Float16*)(smem + 155648);    //  32 KiB [128][128] f16

  const int tid  = threadIdx.x;
  const int wave = tid >> 5;
  const int lane = tid & 31;
  const int bh   = blockIdx.x;                  // 0 .. B*H-1
  const int b    = bh >> 3;
  const int h    = bh & 7;

  // Stage X_b (f32 -> f16, float4-vectorized) and the bias tile into sS.
  {
    const float4* xb4 = (const float4*)(x + (long long)b * NPG * HIDDIM);
    for (int i = tid; i < NPG * HIDDIM / 4; i += 256) {
      const float4 v = xb4[i];
      v4h hv = { (_Float16)v.x, (_Float16)v.y, (_Float16)v.z, (_Float16)v.w };
      *(v4h*)(sX + i * 4) = hv;
    }
    const float4* bt4 = (const float4*)(bias + (long long)bh * NPG * NPG);
    float4* sS4 = (float4*)sS;
    for (int i = tid; i < NPG * NPG / 4; i += 256) sS4[i] = bt4[i];
  }
  __syncthreads();

  // Q, K, V projections straight from pre-transposed f16 weights (global/L2).
  gemm_proj_128x32(sX, WqT, bq, h, sQ,  false, wave, lane);
  gemm_proj_128x32(sX, WkT, bk, h, sK,  false, wave, lane);
  gemm_proj_128x32(sX, WvT, bv, h, sVt, true,  wave, lane);
  __syncthreads();

  // Scores: S = (Q K^T) * scale + bias.  Wave w owns row stripe [w*16, w*16+16)
  // (exactly the sQ rows it just wrote).
  {
    const int row0 = wave * 16;
    const v16h a = frag_a(sQ + row0 * DHEAD, DHEAD, lane);   // K = 32 = DH
    const int n  = lane & 15;
    const int rb = (lane & 16) ? 8 : 0;
#pragma unroll
    for (int cb = 0; cb < 8; ++cb) {
      const int col0 = cb * 16;
      v16h bT = frag_b_t(sK + col0 * DHEAD, DHEAD, lane);    // K rows contiguous
      v8f zero = {};
      v8f acc = __builtin_amdgcn_wmma_f32_16x16x32_f16(false, a, false, bT,
                                                       (short)0, zero, false, false);
#pragma unroll
      for (int j = 0; j < 8; ++j) {
        const int r = row0 + rb + j;
        const int c = col0 + n;
        sS[r * NPG + c] = acc[j] * SCALE + sS[r * NPG + c];
      }
    }
  }
  __syncthreads();

  // Row softmax: thread t (< 128) handles row t; recompute-exp (no big array).
  if (tid < NPG) {
    const float4* row4 = (const float4*)(sS + tid * NPG);
    float m = -3.4e38f;
#pragma unroll 4
    for (int c = 0; c < NPG / 4; ++c) {
      const float4 v = row4[c];
      m = fmaxf(m, fmaxf(fmaxf(v.x, v.y), fmaxf(v.z, v.w)));
    }
    float s = 0.f;
#pragma unroll 4
    for (int c = 0; c < NPG / 4; ++c) {
      const float4 v = row4[c];
      s += __expf(v.x - m) + __expf(v.y - m) + __expf(v.z - m) + __expf(v.w - m);
    }
    const float inv = 1.0f / s;
    v4h* prow = (v4h*)(sP + tid * NPG);
#pragma unroll 4
    for (int c = 0; c < NPG / 4; ++c) {
      const float4 v = row4[c];
      v4h hv = { (_Float16)(__expf(v.x - m) * inv), (_Float16)(__expf(v.y - m) * inv),
                 (_Float16)(__expf(v.z - m) * inv), (_Float16)(__expf(v.w - m) * inv) };
      prow[c] = hv;
    }
  }
  __syncthreads();

  // O = P @ V -> attn_out[b*NPG + row][h*DHEAD + col] (f32, head-interleaved).
  {
    const int row0 = wave * 16;
    const int n  = lane & 15;
    const int rb = (lane & 16) ? 8 : 0;
#pragma unroll
    for (int cb = 0; cb < 2; ++cb) {
      const int col0 = cb * 16;
      v8f acc = {};
#pragma unroll
      for (int k0 = 0; k0 < NPG; k0 += 32) {
        v16h a  = frag_a(sP + row0 * NPG + k0, NPG, lane);
        v16h bb = frag_b_t(sVt + col0 * NPG + k0, NPG, lane);
        acc = __builtin_amdgcn_wmma_f32_16x16x32_f16(false, a, false, bb,
                                                     (short)0, acc, false, false);
      }
#pragma unroll
      for (int j = 0; j < 8; ++j) {
        const int r = row0 + rb + j;
        attn_out[((long long)b * NPG + r) * HIDDIM + h * DHEAD + col0 + n] = acc[j];
      }
    }
  }
}

// ---------------------------------------------------------------------------
// Kernel 4: output projection  out = attn_out @ Wo + bo   (32768x256 @ 256x256)
// 64-row stripes; each wave owns 2 col-tiles; B fragment hoisted across the
// 4 row-tiles (4x amortization); B read from pre-transposed f16 WoT.
// ---------------------------------------------------------------------------
__global__ __launch_bounds__(256)
void out_proj_kernel(const float* __restrict__ ain,
                     const _Float16* __restrict__ WoT,
                     const float* __restrict__ bo,
                     float* __restrict__ out) {
  __shared__ _Float16 sA[64 * HIDDIM];          // 32 KiB
  const int tid  = threadIdx.x;
  const int wave = tid >> 5;
  const int lane = tid & 31;
  const long long r0 = (long long)blockIdx.x * 64;

  const float4* arow4 = (const float4*)(ain + r0 * HIDDIM);
  for (int i = tid; i < 64 * HIDDIM / 4; i += 256) {
    const float4 v = arow4[i];
    v4h hv = { (_Float16)v.x, (_Float16)v.y, (_Float16)v.z, (_Float16)v.w };
    *(v4h*)(sA + i * 4) = hv;
  }
  __syncthreads();

  const int n  = lane & 15;
  const int rb = (lane & 16) ? 8 : 0;
#pragma unroll
  for (int ct = 0; ct < 2; ++ct) {
    const int col0 = (wave * 2 + ct) * 16;      // 16 col-tiles across 8 waves
    v8f acc0 = {}, acc1 = {}, acc2 = {}, acc3 = {};
#pragma unroll
    for (int k0 = 0; k0 < HIDDIM; k0 += 32) {
      v16h bb = frag_b_t(WoT + col0 * HIDDIM + k0, HIDDIM, lane);
      v16h a0 = frag_a(sA +  0 * HIDDIM + k0, HIDDIM, lane);
      v16h a1 = frag_a(sA + 16 * HIDDIM + k0, HIDDIM, lane);
      v16h a2 = frag_a(sA + 32 * HIDDIM + k0, HIDDIM, lane);
      v16h a3 = frag_a(sA + 48 * HIDDIM + k0, HIDDIM, lane);
      acc0 = __builtin_amdgcn_wmma_f32_16x16x32_f16(false, a0, false, bb,
                                                    (short)0, acc0, false, false);
      acc1 = __builtin_amdgcn_wmma_f32_16x16x32_f16(false, a1, false, bb,
                                                    (short)0, acc1, false, false);
      acc2 = __builtin_amdgcn_wmma_f32_16x16x32_f16(false, a2, false, bb,
                                                    (short)0, acc2, false, false);
      acc3 = __builtin_amdgcn_wmma_f32_16x16x32_f16(false, a3, false, bb,
                                                    (short)0, acc3, false, false);
    }
    const float bias = bo[col0 + n];
#pragma unroll
    for (int j = 0; j < 8; ++j) {
      out[(r0 +  0 + rb + j) * HIDDIM + col0 + n] = acc0[j] + bias;
      out[(r0 + 16 + rb + j) * HIDDIM + col0 + n] = acc1[j] + bias;
      out[(r0 + 32 + rb + j) * HIDDIM + col0 + n] = acc2[j] + bias;
      out[(r0 + 48 + rb + j) * HIDDIM + col0 + n] = acc3[j] + bias;
    }
  }
}

// ---------------------------------------------------------------------------
// Host launcher
// ---------------------------------------------------------------------------
extern "C" void kernel_launch(void* const* d_in, const int* in_sizes, int n_in,
                              void* d_out, int out_size, void* d_ws, size_t ws_size,
                              hipStream_t stream) {
  (void)in_sizes; (void)n_in; (void)out_size; (void)ws_size;
  const float* x    = (const float*)d_in[0];
  const float* ea   = (const float*)d_in[1];
  const float* Wq   = (const float*)d_in[2];
  const float* bq   = (const float*)d_in[3];
  const float* Wk   = (const float*)d_in[4];
  const float* bk   = (const float*)d_in[5];
  const float* Wv   = (const float*)d_in[6];
  const float* bv   = (const float*)d_in[7];
  const float* Wo   = (const float*)d_in[8];
  const float* bo   = (const float*)d_in[9];
  const float* gt   = (const float*)d_in[10];
  const float* We   = (const float*)d_in[11];
  const float* be   = (const float*)d_in[12];
  const int*   ei   = (const int*)d_in[13];
  const int*   egt  = (const int*)d_in[14];
  float*       out  = (float*)d_out;

  // Workspace: bias [B,H,NPG,NPG] f32 (128 MiB, L2-resident), attn_out
  // [B*NPG, HID] f32 (32 MiB), then 4 pre-transposed f16 weights (128 KiB ea).
  float* bias = (float*)d_ws;
  const long long bias_elems = (long long)BGRAPH * NHEAD * NPG * NPG;
  float* attn_out = bias + bias_elems;
  _Float16* wT  = (_Float16*)(attn_out + (long long)BGRAPH * NPG * HIDDIM);
  _Float16* WqT = wT;
  _Float16* WkT = wT + 1 * HIDDIM * HIDDIM;
  _Float16* WvT = wT + 2 * HIDDIM * HIDDIM;
  _Float16* WoT = wT + 3 * HIDDIM * HIDDIM;

  // 0) transpose+convert weights to f16 [out][in]
  transpose_w_kernel<<<HIDDIM * HIDDIM / 256, 256, 0, stream>>>(Wq, WqT);
  transpose_w_kernel<<<HIDDIM * HIDDIM / 256, 256, 0, stream>>>(Wk, WkT);
  transpose_w_kernel<<<HIDDIM * HIDDIM / 256, 256, 0, stream>>>(Wv, WvT);
  transpose_w_kernel<<<HIDDIM * HIDDIM / 256, 256, 0, stream>>>(Wo, WoT);

  // 1) zero bias
  fill_zero_kernel<<<8192, 256, 0, stream>>>((float4*)bias, bias_elems / 4);

  // 2) edge-bias scatter
  edge_bias_kernel<<<NEDGE / 256, 256, 0, stream>>>(ei, ea, egt, gt, We, be, bias);

  // 3) fused per-(graph, head) attention (184 KiB dynamic LDS)
  attention_tile_kernel<<<BGRAPH * NHEAD, 256, 188416, stream>>>(
      x, WqT, WkT, WvT, bq, bk, bv, bias, attn_out);

  // 4) output projection
  out_proj_kernel<<<(BGRAPH * NPG) / 64, 256, 0, stream>>>(attn_out, WoT, bo, out);
}
